// LightningAttention_10814727651703
// MI455X (gfx1250) — compile-verified, matrix operand order checked by reading, not run on hardware
//
#include <hip/hip_runtime.h>
#include <hip/hip_bf16.h>
#include <cstdint>

// ---------------------------------------------------------------------------
// LightningAttention for MI455X (gfx1250, wave32, WMMA bf16 16x16x32)
// b=8, l=16384, c=256, h=8, d=32, CHUNK=128, n=128 chunks
//
// All matrix operands are loaded in the "A-style" contiguous per-lane pattern
// (two 16-byte runs per lane -> b128 loads). B operands are therefore kept in
// TRANSPOSED storage: a B fragment B[k][n] is read from Bt[n][k].
// ---------------------------------------------------------------------------

typedef __attribute__((ext_vector_type(16))) __bf16 v16bf;
typedef __attribute__((ext_vector_type(8)))  float  v8f;

#define B_SZ   8
#define L_SZ   16384
#define C_SZ   256
#define H_SZ   8
#define D_SZ   32
#define CHK    128
#define NCHK   128    // L_SZ / CHK
#define BH     64     // B_SZ * H_SZ
#define M_ROWS 131072 // B_SZ * L_SZ
#define QSCALE 0.17677669529663687f  // 32^-0.5

__device__ __forceinline__ int frag_k(int j, int lane) {
    // K offset for slot j (0..15) of a 16-element bf16 fragment (wave32).
    // lanes 0-15 : K in {0..7, 16..23};  lanes 16-31 : K in {8..15, 24..31}
    int khalf = (lane >> 4) & 1;
    return (j & 7) + (khalf << 3) + ((j >> 3) << 4);
}

// Load a 16x32 fragment in A layout from row-major storage (row stride ld).
// Used for A operands (p = A tile, rows = M) and for B operands stored
// transposed (p = Bt tile, rows = N, so element (n,k) = B[k][n]).
// Each lane reads two contiguous 16-byte runs -> b128 loads.
__device__ __forceinline__ v16bf load_frag(const __bf16* p, int ld, int lane) {
    v16bf a;
    const __bf16* rp = p + (size_t)(lane & 15) * ld;
#pragma unroll
    for (int j = 0; j < 16; ++j) a[j] = rp[frag_k(j, lane)];
    return a;
}

__device__ __forceinline__ v8f wmma_bf16(v16bf a, v16bf b, v8f c) {
    return __builtin_amdgcn_wmma_f32_16x16x32_bf16(
        /*neg_a=*/false, a, /*neg_b=*/false, b,
        /*c_mod=*/(short)0, c, /*reuse_a=*/false, /*reuse_b=*/false);
}

// pack 8 fp32 (two float4) into 8 bf16 and return as uint4 for a b128 store
__device__ __forceinline__ uint4 pack8_bf16(float4 f0, float4 f1) {
    union { __bf16 b[8]; uint4 u; } pk;
    pk.b[0] = (__bf16)f0.x; pk.b[1] = (__bf16)f0.y;
    pk.b[2] = (__bf16)f0.z; pk.b[3] = (__bf16)f0.w;
    pk.b[4] = (__bf16)f1.x; pk.b[5] = (__bf16)f1.y;
    pk.b[6] = (__bf16)f1.z; pk.b[7] = (__bf16)f1.w;
    return pk.u;
}

// ---------------------------------------------------------------------------
// Kernel 1: qkv = x @ w_qkv ; split into q (scaled, (bh,l,d)), k ((bh,l,d)),
// vT ((bh,n,d,C)), all bf16.  Block tile 128x128, K=256 in 8 steps of 32.
// grid (6 N-tiles, 1024 M-tiles) -> N fastest so the 128-row slab of x is
// L2-resident across its 6 N-blocks.  block 256 (8 waves as 4(M) x 2(N)).
// ---------------------------------------------------------------------------
__global__ __launch_bounds__(256)
void la_qkv_kernel(const float* __restrict__ x, const float* __restrict__ wqkv,
                   __bf16* __restrict__ qw, __bf16* __restrict__ kw,
                   __bf16* __restrict__ vTw)
{
    __shared__ __align__(16) __bf16 As[128 * 32];   // (m, k)
    __shared__ __align__(16) __bf16 BsT[128 * 32];  // (n, k)  transposed
    const int tid = threadIdx.x, lane = tid & 31, w = tid >> 5;
    const int n0 = blockIdx.x * 128;
    const int m0 = blockIdx.y * 128;
    const int wm = w >> 1, wn = w & 1;

    v8f acc[2][4];
#pragma unroll
    for (int ms = 0; ms < 2; ++ms)
#pragma unroll
        for (int ns = 0; ns < 4; ++ns) acc[ms][ns] = (v8f){};

    for (int k0 = 0; k0 < 256; k0 += 32) {
        // stage A tile (128x32) fp32 -> bf16: 8 consecutive floats per step
#pragma unroll
        for (int o = tid; o < 512; o += 256) {
            int r = o >> 2, c8 = (o & 3) * 8;
            const float4* src = (const float4*)&x[(size_t)(m0 + r) * 256 + k0 + c8];
            *((uint4*)(As + r * 32 + c8)) = pack8_bf16(src[0], src[1]);
        }
        // stage B tile (32x128) fp32 -> bf16, TRANSPOSED to (n, k)
#pragma unroll
        for (int o = tid; o < 1024; o += 256) {
            int r = o >> 5, c4 = (o & 31) * 4;   // r = k row (coalesced read)
            float4 f = *(const float4*)&wqkv[(size_t)(k0 + r) * 768 + n0 + c4];
            BsT[(c4 + 0) * 32 + r] = (__bf16)f.x;
            BsT[(c4 + 1) * 32 + r] = (__bf16)f.y;
            BsT[(c4 + 2) * 32 + r] = (__bf16)f.z;
            BsT[(c4 + 3) * 32 + r] = (__bf16)f.w;
        }
        if (k0 + 32 < 256) {
            __builtin_prefetch(&x[(size_t)(m0 + (tid >> 1)) * 256 + k0 + 32], 0, 1);
            __builtin_prefetch(&wqkv[(size_t)(k0 + 32 + (tid >> 4)) * 768 + n0 + (tid & 15) * 8], 0, 1);
        }
        __syncthreads();

        v16bf af[2];
#pragma unroll
        for (int ms = 0; ms < 2; ++ms)
            af[ms] = load_frag(As + (size_t)(wm * 32 + ms * 16) * 32, 32, lane);
#pragma unroll
        for (int ns = 0; ns < 4; ++ns) {
            v16bf bf = load_frag(BsT + (size_t)(wn * 64 + ns * 16) * 32, 32, lane);
#pragma unroll
            for (int ms = 0; ms < 2; ++ms)
                acc[ms][ns] = wmma_bf16(af[ms], bf, acc[ms][ns]);
        }
        __syncthreads();
    }

    // epilogue: scatter into q / k / vT (each 16-col tile lies in one comp+head)
#pragma unroll
    for (int ms = 0; ms < 2; ++ms) {
#pragma unroll
        for (int ns = 0; ns < 4; ++ns) {
            int gr = m0 + wm * 32 + ms * 16;
            int gc = n0 + wn * 64 + ns * 16;
            int comp = gc >> 8;              // 0=q 1=k 2=v
            int head = (gc & 255) >> 5;
            int Cj = gc + (lane & 15);
            int dd = Cj & 31;
#pragma unroll
            for (int r = 0; r < 8; ++r) {
                int R = gr + r + ((lane >> 4) << 3);
                int bb = R >> 14, t = R & 16383;
                int bh = bb * 8 + head;
                float val = acc[ms][ns][r];
                if (comp == 0) {
                    qw[((size_t)bh * L_SZ + t) * 32 + dd] = (__bf16)(val * QSCALE);
                } else if (comp == 1) {
                    kw[((size_t)bh * L_SZ + t) * 32 + dd] = (__bf16)val;
                } else {
                    int ch = t >> 7, tc = t & 127;
                    vTw[(((size_t)bh * NCHK + ch) * 32 + dd) * 128 + tc] = (__bf16)val;
                }
            }
        }
    }
}

// ---------------------------------------------------------------------------
// Kernel 2a: per-chunk state increment  dS = (k * k_dec)^T @ v  (32x32 fp32)
// grid (128 chunks, 64 bh), block 128 (4 waves, each one 16x16 output tile).
// A (= k^T, rows d, K = position) is transposed through an LDS stage of the
// k chunk; B (= v) reads contiguously from vT.
// ---------------------------------------------------------------------------
__global__ __launch_bounds__(128)
void la_delta_kernel(const __bf16* __restrict__ kw, const __bf16* __restrict__ vTw,
                     const float* __restrict__ slopes, float* __restrict__ Sdelta)
{
    __shared__ __align__(16) __bf16 ks[CHK * 32];  // k chunk, (tc, dd)
    __shared__ float dec[CHK + 1];                 // dec[j] = exp(-s*j)
    const int ch = blockIdx.x, bh = blockIdx.y;
    const float s = slopes[bh & 7];
    const int tid = threadIdx.x, lane = tid & 31, w = tid >> 5;
    const int mt = w >> 1, nt = w & 1;

    const __bf16* kc  = kw  + ((size_t)bh * L_SZ + (size_t)ch * CHK) * 32;
    const __bf16* vTc = vTw + ((size_t)bh * NCHK + ch) * 4096;

    for (int i = tid; i < CHK + 1; i += 128) dec[i] = __expf(-s * (float)i);
    // stage k chunk: 128x32 bf16 = 512 uint4
#pragma unroll
    for (int e = tid; e < 512; e += 128)
        ((uint4*)ks)[e] = ((const uint4*)kc)[e];
    __syncthreads();

    v8f acc = (v8f){};
    const int mrow = mt * 16 + (lane & 15);        // A row = d index
#pragma unroll
    for (int kt = 0; kt < 4; ++kt) {
        v16bf a;
#pragma unroll
        for (int j = 0; j < 16; ++j) {
            int kg = kt * 32 + frag_k(j, lane);    // position within chunk
            a[j] = (__bf16)((float)ks[kg * 32 + mrow] * dec[CHK - kg]);
        }
        v16bf bv = load_frag(vTc + (size_t)(nt * 16) * 128 + kt * 32, 128, lane);
        acc = wmma_bf16(a, bv, acc);
    }

    float* out = Sdelta + ((size_t)bh * NCHK + ch) * 1024;
#pragma unroll
    for (int r = 0; r < 8; ++r) {
        int drow = mt * 16 + r + ((lane >> 4) << 3);
        int e = nt * 16 + (lane & 15);
        out[drow * 32 + e] = acc[r];               // linear S[d][e]
    }
}

// ---------------------------------------------------------------------------
// Kernel 2b: decayed prefix scan over chunks (the only serial part; trivial).
// SpreT[ch] = state BEFORE chunk ch, stored TRANSPOSED (e, d) in bf16 so the
// inter-chunk B fragments load contiguously.
// grid 64, block 256; each thread owns 4 of the 1024 state elements.
// ---------------------------------------------------------------------------
__global__ __launch_bounds__(256)
void la_scan_kernel(const float* __restrict__ Sdelta, const float* __restrict__ slopes,
                    __bf16* __restrict__ SpreT)
{
    const int bh = blockIdx.x;
    const float blk = __expf(-slopes[bh & 7] * (float)CHK);
    const int tid = threadIdx.x;
    const float* din = Sdelta + (size_t)bh * NCHK * 1024;
    __bf16* dout = SpreT + (size_t)bh * NCHK * 1024;
    float run[4] = {0.f, 0.f, 0.f, 0.f};
    for (int ch = 0; ch < NCHK; ++ch) {
#pragma unroll
        for (int i = 0; i < 4; ++i) {
            int e = tid + i * 256;                 // linear S index: d = e>>5, e2 = e&31
            dout[(size_t)ch * 1024 + (e & 31) * 32 + (e >> 5)] = (__bf16)run[i];
            run[i] = blk * run[i] + din[(size_t)ch * 1024 + e];
        }
    }
}

// ---------------------------------------------------------------------------
// Kernel 2c: per-chunk output = mask(q k^T) v + (q_dec*q) S_pre, bf16 out in
// (b, l, c) layout for the projection GEMM.
// grid (128 chunks, 64 bh), block 256 (8 waves, each 16 rows of the chunk).
// Every operand fragment load is contiguous: q (A), k (Bt), scores-LDS (A),
// vT (Bt), SpreT (Bt).
// ---------------------------------------------------------------------------
__global__ __launch_bounds__(256)
void la_attn_kernel(const __bf16* __restrict__ qw, const __bf16* __restrict__ kw,
                    const __bf16* __restrict__ vTw, const __bf16* __restrict__ SpreT,
                    const float* __restrict__ slopes, __bf16* __restrict__ ow)
{
    __shared__ __align__(16) __bf16 sc[CHK * CHK]; // masked scores, (t, i) bf16
    __shared__ float dec[CHK + 1];                 // dec[j] = exp(-s*j)
    const int ch = blockIdx.x, bh = blockIdx.y;
    const int bb = bh >> 3, hd = bh & 7;
    const float s = slopes[hd];
    const int tid = threadIdx.x, lane = tid & 31, w = tid >> 5;

    for (int i = tid; i < CHK + 1; i += 256) dec[i] = __expf(-s * (float)i);
    __syncthreads();

    const __bf16* qc  = qw   + ((size_t)bh * L_SZ + (size_t)ch * CHK) * 32;
    const __bf16* kc  = kw   + ((size_t)bh * L_SZ + (size_t)ch * CHK) * 32;
    const __bf16* vTc = vTw  + ((size_t)bh * NCHK + ch) * 4096;
    const __bf16* Sct = SpreT + ((size_t)bh * NCHK + ch) * 1024;

    const int r0 = w * 16;
    v16bf aq = load_frag(qc + (size_t)r0 * 32, 32, lane);

    // scores = q k^T, masked+decayed, into LDS as bf16
    v8f zacc = (v8f){};
#pragma unroll
    for (int nt = 0; nt < 8; ++nt) {
        v16bf bk = load_frag(kc + (size_t)(nt * 16) * 32, 32, lane);
        v8f accs = wmma_bf16(aq, bk, zacc);
        int i = nt * 16 + (lane & 15);
#pragma unroll
        for (int r = 0; r < 8; ++r) {
            int t = r0 + r + ((lane >> 4) << 3);
            float mv = (t >= i) ? dec[t - i] : 0.0f;
            sc[t * 128 + i] = (__bf16)(accs[r] * mv);
        }
    }
    __syncthreads();

    // inter-chunk: (q * q_dec) @ S_pre   (B from transposed SpreT)
    const float fq = dec[r0 + (lane & 15)];
    v16bf aqd;
#pragma unroll
    for (int j = 0; j < 16; ++j) aqd[j] = (__bf16)((float)aq[j] * fq);

    v8f oacc[2];
#pragma unroll
    for (int nt = 0; nt < 2; ++nt) {
        v16bf bs = load_frag(Sct + (size_t)(nt * 16) * 32, 32, lane);
        oacc[nt] = wmma_bf16(aqd, bs, zacc);
    }

    // intra-chunk: scores @ v   (A from LDS, B from vT)
#pragma unroll
    for (int nt = 0; nt < 2; ++nt) {
#pragma unroll
        for (int kt = 0; kt < 4; ++kt) {
            v16bf as = load_frag(sc + (size_t)r0 * 128 + kt * 32, 128, lane);
            v16bf bv = load_frag(vTc + (size_t)(nt * 16) * 128 + kt * 32, 128, lane);
            oacc[nt] = wmma_bf16(as, bv, oacc[nt]);
        }
    }

    // write o in (b, l, c) layout
#pragma unroll
    for (int nt = 0; nt < 2; ++nt) {
#pragma unroll
        for (int r = 0; r < 8; ++r) {
            int t = ch * CHK + r0 + r + ((lane >> 4) << 3);
            int col = hd * 32 + nt * 16 + (lane & 15);
            ow[((size_t)bb * L_SZ + t) * 256 + col] = (__bf16)oacc[nt][r];
        }
    }
}

// ---------------------------------------------------------------------------
// Kernel 3: out = o @ w_proj + b_proj (fp32 out). Same tiling as kernel 1.
// grid (2 N-tiles, 1024 M-tiles) -> o slab L2-resident across its N-blocks.
// ---------------------------------------------------------------------------
__global__ __launch_bounds__(256)
void la_proj_kernel(const __bf16* __restrict__ ow, const float* __restrict__ wproj,
                    const float* __restrict__ bproj, float* __restrict__ out)
{
    __shared__ __align__(16) __bf16 As[128 * 32];   // (m, k)
    __shared__ __align__(16) __bf16 BsT[128 * 32];  // (n, k)  transposed
    const int tid = threadIdx.x, lane = tid & 31, w = tid >> 5;
    const int n0 = blockIdx.x * 128;
    const int m0 = blockIdx.y * 128;
    const int wm = w >> 1, wn = w & 1;

    v8f acc[2][4];
#pragma unroll
    for (int ms = 0; ms < 2; ++ms)
#pragma unroll
        for (int ns = 0; ns < 4; ++ns) acc[ms][ns] = (v8f){};

    for (int k0 = 0; k0 < 256; k0 += 32) {
        // stage A tile (128x32) bf16 -> straight uint4 copy (row = 4 uint4)
#pragma unroll
        for (int o = tid; o < 512; o += 256) {
            int r = o >> 2, q = (o & 3) * 8;
            ((uint4*)As)[o] = *(const uint4*)&ow[(size_t)(m0 + r) * 256 + k0 + q];
        }
        // stage B tile (32x128) fp32 -> bf16, TRANSPOSED to (n, k)
#pragma unroll
        for (int o = tid; o < 1024; o += 256) {
            int r = o >> 5, c4 = (o & 31) * 4;
            float4 f = *(const float4*)&wproj[(size_t)(k0 + r) * 256 + n0 + c4];
            BsT[(c4 + 0) * 32 + r] = (__bf16)f.x;
            BsT[(c4 + 1) * 32 + r] = (__bf16)f.y;
            BsT[(c4 + 2) * 32 + r] = (__bf16)f.z;
            BsT[(c4 + 3) * 32 + r] = (__bf16)f.w;
        }
        if (k0 + 32 < 256) {
            __builtin_prefetch(&ow[(size_t)(m0 + (tid >> 1)) * 256 + k0 + 32], 0, 1);
        }
        __syncthreads();

        v16bf af[2];
#pragma unroll
        for (int ms = 0; ms < 2; ++ms)
            af[ms] = load_frag(As + (size_t)(wm * 32 + ms * 16) * 32, 32, lane);
#pragma unroll
        for (int ns = 0; ns < 4; ++ns) {
            v16bf bf = load_frag(BsT + (size_t)(wn * 64 + ns * 16) * 32, 32, lane);
#pragma unroll
            for (int ms = 0; ms < 2; ++ms)
                acc[ms][ns] = wmma_bf16(af[ms], bf, acc[ms][ns]);
        }
        __syncthreads();
    }

#pragma unroll
    for (int ms = 0; ms < 2; ++ms) {
#pragma unroll
        for (int ns = 0; ns < 4; ++ns) {
            int gr = m0 + wm * 32 + ms * 16;
            int Cj = n0 + wn * 64 + ns * 16 + (lane & 15);
            float bias = bproj[Cj];
#pragma unroll
            for (int r = 0; r < 8; ++r) {
                int R = gr + r + ((lane >> 4) << 3);
                out[(size_t)R * 256 + Cj] = acc[ms][ns][r] + bias;
            }
        }
    }
}

// ---------------------------------------------------------------------------
// Launch
// ---------------------------------------------------------------------------
extern "C" void kernel_launch(void* const* d_in, const int* in_sizes, int n_in,
                              void* d_out, int out_size, void* d_ws, size_t ws_size,
                              hipStream_t stream) {
    const float* x      = (const float*)d_in[0];
    const float* wqkv   = (const float*)d_in[1];
    const float* wproj  = (const float*)d_in[2];
    const float* bproj  = (const float*)d_in[3];
    const float* slopes = (const float*)d_in[4];

    // workspace layout (bytes)
    const size_t SZ_T = (size_t)BH * L_SZ * D_SZ * sizeof(__bf16); // 67,108,864
    char* ws = (char*)d_ws;
    __bf16* qw   = (__bf16*)(ws);
    __bf16* kw   = (__bf16*)(ws + SZ_T);
    __bf16* vTw  = (__bf16*)(ws + 2 * SZ_T);
    __bf16* owb  = (__bf16*)(ws + 3 * SZ_T);
    float*  Sdel = (float*) (ws + 4 * SZ_T);
    __bf16* Spre = (__bf16*)(ws + 4 * SZ_T + (size_t)BH * NCHK * 1024 * sizeof(float));

    la_qkv_kernel<<<dim3(6, M_ROWS / 128), 256, 0, stream>>>(x, wqkv, qw, kw, vTw);
    la_delta_kernel<<<dim3(NCHK, BH), 128, 0, stream>>>(kw, vTw, slopes, Sdel);
    la_scan_kernel<<<dim3(BH), 256, 0, stream>>>(Sdel, slopes, Spre);
    la_attn_kernel<<<dim3(NCHK, BH), 256, 0, stream>>>(qw, kw, vTw, Spre, slopes, owb);
    la_proj_kernel<<<dim3(2, M_ROWS / 128), 256, 0, stream>>>(owb, wproj, bproj, (float*)d_out);
}
